// SequenceModel_16441134809491
// MI455X (gfx1250) — matrix operand order repaired and checked by reading, not compiled, not run on hardware
//
#include <hip/hip_runtime.h>
#include <math.h>

// ---------------------------------------------------------------------------
// Latent ODE, RK4 x191, f(y)=tanh(y@W1+b1)@W2+b2, M=128, 3200<->4096.
// Compute-bound (~5.1 TFLOP); weights fit L2 as bf16 -> bf16 WMMA GEMM
// (v_wmma_f32_16x16x32_bf16) with f32 accumulate, fused RK4 epilogues.
// Tile staging now uses GLOBAL_LOAD_ASYNC_TO_LDS_B128 (ASYNCcnt) so the
// DMA-style copy overlaps the WMMA stream with no VGPR transit.
// ---------------------------------------------------------------------------

typedef __bf16 bf16x16 __attribute__((ext_vector_type(16)));
typedef float  f32x8   __attribute__((ext_vector_type(8)));

__device__ __forceinline__ unsigned short f2bf(float f) {
  unsigned int u = __float_as_uint(f);
  u += 0x7FFFu + ((u >> 16) & 1u);          // round-to-nearest-even
  return (unsigned short)(u >> 16);
}

// issue one per-lane 16B async global->LDS transfer (GVS addressing)
__device__ __forceinline__ void async_b128(unsigned ldsByteAddr,
                                           unsigned byteOff,
                                           const void* base) {
  asm volatile("global_load_async_to_lds_b128 %0, %1, %2"
               :: "v"(ldsByteAddr), "v"(byteOff), "s"(base)
               : "memory");
}
__device__ __forceinline__ void wait_async0() {
  asm volatile("s_wait_asynccnt 0x0" ::: "memory");
}

// -------------------- weight convert + transpose (f32 KxN -> bf16 NxK) ------
__global__ void k_convT(const float* __restrict__ W, unsigned short* __restrict__ Wt,
                        int K, int N) {
  long long i = (long long)blockIdx.x * blockDim.x + threadIdx.x;
  long long total = (long long)K * N;
  if (i >= total) return;
  int n = (int)(i / K);
  int k = (int)(i % K);
  Wt[i] = f2bf(W[(long long)k * N + n]);
}

// -------------------- prologue (tiny: ~0.3 GFLOP total) ---------------------
__global__ void k_dec(const float* __restrict__ z, const float* __restrict__ W,
                      const float* __restrict__ b, float* __restrict__ E) {
  int i = blockIdx.x * blockDim.x + threadIdx.x;
  if (i >= 128 * 2048) return;
  int bi = i >> 11, j = i & 2047;
  float s = b[j];
  const float* zr = z + bi * 128;
  for (int k = 0; k < 128; ++k) s += zr[k] * W[k * 2048 + j];
  E[i] = s;
}

__global__ void k_def1(const float* __restrict__ E, const float* __restrict__ W1,
                       const float* __restrict__ b1, float* __restrict__ H) {
  int i = blockIdx.x * blockDim.x + threadIdx.x;
  if (i >= 4096 * 128) return;
  int s = i >> 7, j = i & 127;
  float v = b1[j];
  const float* e = E + s * 64;
  for (int k = 0; k < 64; ++k) v += e[k] * W1[k * 128 + j];
  H[i] = v > 0.f ? v : 0.f;
}

__global__ void k_def2(const float* __restrict__ H, const float* __restrict__ W2,
                       const float* __restrict__ b2, const float* __restrict__ T,
                       float* __restrict__ C) {
  int i = blockIdx.x * blockDim.x + threadIdx.x;
  if (i >= 4096 * 99) return;
  int s = i / 99, c = i % 99;
  float v = b2[c] + T[(s & 31) * 99 + c];
  const float* h = H + s * 128;
  for (int k = 0; k < 128; ++k) v += h[k] * W2[k * 99 + c];
  C[i] = v;
}

__global__ void k_alive(const float* __restrict__ E, const float* __restrict__ W1,
                        const float* __restrict__ b1, const float* __restrict__ W2,
                        const float* __restrict__ b2, float* __restrict__ alive) {
  int s = blockIdx.x * blockDim.x + threadIdx.x;
  if (s >= 4096) return;
  const float* e = E + s * 64;
  float logit = b2[0];
  for (int j = 0; j < 32; ++j) {
    float v = b1[j];
    for (int k = 0; k < 64; ++k) v += e[k] * W1[k * 32 + j];
    v = v > 0.f ? v : 0.f;
    logit += v * W2[j];
  }
  alive[s] = 1.0f / (1.0f + __expf(-logit));
}

__global__ void k_y0(const float* __restrict__ C, const float* __restrict__ alive,
                     float* __restrict__ y0, unsigned short* __restrict__ ybf) {
  int i = blockIdx.x * blockDim.x + threadIdx.x;
  if (i >= 128 * 3200) return;
  int b = i / 3200, c = i % 3200;
  float v;
  if (c < 3168) {
    int s = c / 99, cc = c % 99;
    v = C[(b * 32 + s) * 99 + cc] * alive[b * 32 + s];
  } else {
    v = alive[b * 32 + (c - 3168)];
  }
  y0[i] = v;
  ybf[i] = f2bf(v);
}

// -------------------- WMMA GEMM with fused RK4 epilogues --------------------
// A: MxK bf16 row-major (activations); Bt: NxK bf16 row-major (transposed wt).
// MODE 0: out = bf16(tanh(v))                                  (hidden layer)
// MODE 1: acc = v;       t = y + cT*v -> bf16                  (k1)
// MODE 2: acc += accW*v; t = y + cT*v -> bf16                  (k2,k3)
// MODE 3: yn = y + dt6*(acc+v); yOut = yn; out = bf16(yn)      (k4 + update)
#define BM 64
#define BN 64
#define BK 32
#define LDSS 40   // shorts per LDS row (80 B: keeps b128 transfers 16B-aligned)

template <int MODE>
__launch_bounds__(128)
__global__ void k_gemm(const unsigned short* __restrict__ A,
                       const unsigned short* __restrict__ Bt,
                       const float* __restrict__ bias,
                       int N, int K,
                       unsigned short* __restrict__ outBf,
                       const float* __restrict__ yIn,
                       float* __restrict__ acc,
                       float* __restrict__ yOut,
                       float accW, float cT, float dt6) {
  __shared__ __align__(16) unsigned short sA[2][BM * LDSS];
  __shared__ __align__(16) unsigned short sB[2][BN * LDSS];

  const int tid = threadIdx.x;
  const int lane = tid & 31;
  const int wave = tid >> 5;
  const int wm = wave >> 1, wn = wave & 1;    // 2x2 wave grid, 32x32 each
  const int lm = lane & 15, lh = lane >> 4;

  const int m0 = blockIdx.y * BM;
  const int n0 = blockIdx.x * BN;

  // tile staging: 64 rows x 32 cols bf16; each thread: 1 row-half (2x16B)
  const int gr = tid >> 1;
  const int gh = tid & 1;
  const unsigned offA = (unsigned)(((m0 + gr) * K + gh * 16) * 2);  // bytes
  const unsigned offB = (unsigned)(((n0 + gr) * K + gh * 16) * 2);  // bytes
  const unsigned ldsRow = (unsigned)((gr * LDSS + gh * 16) * 2);    // bytes
  const unsigned ldsA0 = (unsigned)(unsigned long long)(&sA[0][0]) + ldsRow;
  const unsigned ldsB0 = (unsigned)(unsigned long long)(&sB[0][0]) + ldsRow;
  const unsigned bufStepA = (unsigned)(BM * LDSS * 2);
  const unsigned bufStepB = (unsigned)(BN * LDSS * 2);

  union UA { bf16x16 v; unsigned int u[8]; };

  f32x8 c00 = {0,0,0,0,0,0,0,0}, c01 = {0,0,0,0,0,0,0,0};
  f32x8 c10 = {0,0,0,0,0,0,0,0}, c11 = {0,0,0,0,0,0,0,0};

  const int T = K / BK;

  // async-stage tile 0 into buffer 0 (4 x 16B per thread)
  {
    const unsigned go = 0;
    async_b128(ldsA0,      offA + go,      A);
    async_b128(ldsA0 + 16, offA + go + 16, A);
    async_b128(ldsB0,      offB + go,      Bt);
    async_b128(ldsB0 + 16, offB + go + 16, Bt);
  }
  wait_async0();
  __syncthreads();

  int buf = 0;
  for (int t = 0; t < T; ++t) {
    const bool more = (t + 1 < T);
    if (more) {
      // issue next tile into the back buffer; overlaps with WMMA below
      const unsigned go = (unsigned)((t + 1) * BK * 2);
      const unsigned bsel = (unsigned)(buf ^ 1);
      async_b128(ldsA0 + bsel * bufStepA,      offA + go,      A);
      async_b128(ldsA0 + bsel * bufStepA + 16, offA + go + 16, A);
      async_b128(ldsB0 + bsel * bufStepB,      offB + go,      Bt);
      async_b128(ldsB0 + bsel * bufStepB + 16, offB + go + 16, Bt);
    }

    // ---- fragment gathers (contiguous k = 16*lh + s for A and B), then 4 WMMAs ----
    {
      const unsigned int* pa = (const unsigned int*)(&sA[buf][0]);
      const unsigned int* pb = (const unsigned int*)(&sB[buf][0]);
      UA a0, a1, b0, b1;
      {
        const unsigned int* rp0 = pa + (wm * 32 + 0 * 16 + lm) * (LDSS / 2) + lh * 8;
        const unsigned int* rp1 = pa + (wm * 32 + 1 * 16 + lm) * (LDSS / 2) + lh * 8;
#pragma unroll
        for (int j = 0; j < 8; ++j) {
          const int d = j;                            // contiguous k = 16*lh + 2j, 2j+1
          a0.u[j] = rp0[d];
          a1.u[j] = rp1[d];
        }
      }
      {
        const unsigned int* rp0 = pb + (wn * 32 + 0 * 16 + lm) * (LDSS / 2) + lh * 8;
        const unsigned int* rp1 = pb + (wn * 32 + 1 * 16 + lm) * (LDSS / 2) + lh * 8;
#pragma unroll
        for (int j = 0; j < 8; ++j) {
          b0.u[j] = rp0[j];
          b1.u[j] = rp1[j];
        }
      }
      c00 = __builtin_amdgcn_wmma_f32_16x16x32_bf16(false, a0.v, false, b0.v, (short)0, c00, false, false);
      c01 = __builtin_amdgcn_wmma_f32_16x16x32_bf16(false, a0.v, false, b1.v, (short)0, c01, false, false);
      c10 = __builtin_amdgcn_wmma_f32_16x16x32_bf16(false, a1.v, false, b0.v, (short)0, c10, false, false);
      c11 = __builtin_amdgcn_wmma_f32_16x16x32_bf16(false, a1.v, false, b1.v, (short)0, c11, false, false);
    }

    if (more) {
      wait_async0();      // this wave's next-tile transfers are in LDS
      __syncthreads();    // every wave's transfers are in LDS
      buf ^= 1;
    }
  }

  // ---- epilogue: C/D layout n=lane%16, m=r+8*(lane/16) ----
#pragma unroll
  for (int fm = 0; fm < 2; ++fm) {
#pragma unroll
    for (int fn = 0; fn < 2; ++fn) {
      const f32x8 cc = (fm == 0) ? (fn == 0 ? c00 : c01) : (fn == 0 ? c10 : c11);
      const int mB = m0 + wm * 32 + fm * 16 + 8 * lh;
      const int nB = n0 + wn * 32 + fn * 16 + lm;
      const float bv = bias[nB];
#pragma unroll
      for (int r = 0; r < 8; ++r) {
        const int m = mB + r;
        const long long idx = (long long)m * N + nB;
        const float v = cc[r] + bv;
        if (MODE == 0) {
          outBf[idx] = f2bf(tanhf(v));
        } else if (MODE == 1) {
          acc[idx] = v;
          outBf[idx] = f2bf(yIn[idx] + cT * v);
        } else if (MODE == 2) {
          acc[idx] += accW * v;
          outBf[idx] = f2bf(yIn[idx] + cT * v);
        } else {
          const float yn = yIn[idx] + dt6 * (acc[idx] + v);
          yOut[idx] = yn;
          outBf[idx] = f2bf(yn);
        }
      }
    }
  }
}

// ---------------------------------------------------------------------------
extern "C" void kernel_launch(void* const* d_in, const int* in_sizes, int n_in,
                              void* d_out, int out_size, void* d_ws, size_t ws_size,
                              hipStream_t stream) {
  const float* z      = (const float*)d_in[0];
  const float* dec_W  = (const float*)d_in[1];
  const float* dec_b  = (const float*)d_in[2];
  const float* templ  = (const float*)d_in[3];
  const float* def_W1 = (const float*)d_in[4];
  const float* def_b1 = (const float*)d_in[5];
  const float* def_W2 = (const float*)d_in[6];
  const float* def_b2 = (const float*)d_in[7];
  const float* al_W1  = (const float*)d_in[8];
  const float* al_b1  = (const float*)d_in[9];
  const float* al_W2  = (const float*)d_in[10];
  const float* al_b2  = (const float*)d_in[11];
  const float* ode_W1 = (const float*)d_in[12];
  const float* ode_b1 = (const float*)d_in[13];
  const float* ode_W2 = (const float*)d_in[14];
  const float* ode_b2 = (const float*)d_in[15];
  float* out = (float*)d_out;                // (192, 128, 3200) f32; slab s = y_s

  // ---- workspace carve (~62 MB) ----
  char* p = (char*)d_ws;
  auto carve = [&](size_t bytes) -> char* {
    char* r = p;
    p += (bytes + 255) & ~(size_t)255;
    return r;
  };
  unsigned short* W1t  = (unsigned short*)carve(4096ull * 3200 * 2); // N1 x K1
  unsigned short* W2t  = (unsigned short*)carve(3200ull * 4096 * 2); // N2 x K2
  unsigned short* Hbf  = (unsigned short*)carve(128ull * 4096 * 2);
  unsigned short* ybf  = (unsigned short*)carve(128ull * 3200 * 2);
  unsigned short* tbf  = (unsigned short*)carve(128ull * 3200 * 2);
  float*          accB = (float*)carve(128ull * 3200 * 4);
  float*          slotE = (float*)carve(128ull * 2048 * 4);
  float*          hdef  = (float*)carve(4096ull * 128 * 4);
  float*          bcrf  = (float*)carve(4096ull * 99 * 4);
  float*          alive = (float*)carve(4096ull * 4);

  // ---- weights -> bf16 transposed (hits HBM once; stays in L2 after) ----
  {
    long long tot = 3200ll * 4096;
    int thr = 256;
    long long blk = (tot + thr - 1) / thr;
    k_convT<<<(unsigned)blk, thr, 0, stream>>>(ode_W1, W1t, 3200, 4096);
    k_convT<<<(unsigned)blk, thr, 0, stream>>>(ode_W2, W2t, 4096, 3200);
  }

  // ---- prologue -> y0 (slab 0) + ybf ----
  k_dec  <<<(128 * 2048 + 255) / 256, 256, 0, stream>>>(z, dec_W, dec_b, slotE);
  k_def1 <<<(4096 * 128 + 255) / 256, 256, 0, stream>>>(slotE, def_W1, def_b1, hdef);
  k_def2 <<<(4096 * 99 + 255) / 256, 256, 0, stream>>>(hdef, def_W2, def_b2, templ, bcrf);
  k_alive<<<(4096 + 127) / 128, 128, 0, stream>>>(slotE, al_W1, al_b1, al_W2, al_b2, alive);
  k_y0   <<<(128 * 3200 + 255) / 256, 256, 0, stream>>>(bcrf, alive, out, ybf);

  // ---- RK4 main loop: 8 WMMA GEMMs per step ----
  const float dt  = 8.0f / 191.0f;
  const float hdt = 0.5f * dt;
  const float dt6 = dt / 6.0f;
  const long long S = 128ll * 3200;
  const dim3 g1(4096 / BN, 128 / BM);  // hidden GEMM: N=4096, K=3200
  const dim3 g2(3200 / BN, 128 / BM);  // output GEMM: N=3200, K=4096

  for (int s = 0; s < 191; ++s) {
    const float* ycur = out + (long long)s * S;
    float* ynext = out + (long long)(s + 1) * S;
    // k1
    k_gemm<0><<<g1, 128, 0, stream>>>(ybf, W1t, ode_b1, 4096, 3200, Hbf, nullptr, nullptr, nullptr, 0.f, 0.f, 0.f);
    k_gemm<1><<<g2, 128, 0, stream>>>(Hbf, W2t, ode_b2, 3200, 4096, tbf, ycur, accB, nullptr, 0.f, hdt, 0.f);
    // k2
    k_gemm<0><<<g1, 128, 0, stream>>>(tbf, W1t, ode_b1, 4096, 3200, Hbf, nullptr, nullptr, nullptr, 0.f, 0.f, 0.f);
    k_gemm<2><<<g2, 128, 0, stream>>>(Hbf, W2t, ode_b2, 3200, 4096, tbf, ycur, accB, nullptr, 2.f, hdt, 0.f);
    // k3
    k_gemm<0><<<g1, 128, 0, stream>>>(tbf, W1t, ode_b1, 4096, 3200, Hbf, nullptr, nullptr, nullptr, 0.f, 0.f, 0.f);
    k_gemm<2><<<g2, 128, 0, stream>>>(Hbf, W2t, ode_b2, 3200, 4096, tbf, ycur, accB, nullptr, 2.f, dt, 0.f);
    // k4 + state update -> slab s+1 (+ bf16 copy for next step)
    k_gemm<0><<<g1, 128, 0, stream>>>(tbf, W1t, ode_b1, 4096, 3200, Hbf, nullptr, nullptr, nullptr, 0.f, 0.f, 0.f);
    k_gemm<3><<<g2, 128, 0, stream>>>(Hbf, W2t, ode_b2, 3200, 4096, ybf, ycur, accB, ynext, 0.f, 0.f, dt6);
  }
}
